// SelfAttention_v2_54159537602870
// MI455X (gfx1250) — matrix operand order repaired and checked by reading, not compiled, not run on hardware
//
#include <hip/hip_runtime.h>

typedef __bf16 bf16;
typedef __attribute__((ext_vector_type(16))) bf16  v16bf;
typedef __attribute__((ext_vector_type(8)))  bf16  v8bf;
typedef __attribute__((ext_vector_type(4)))  bf16  v4bf;
typedef __attribute__((ext_vector_type(8)))  float v8f;
typedef __attribute__((ext_vector_type(4)))  float v4f;

#define SEQ    4096
#define DMODEL 1024

static __device__ __forceinline__ v16bf cat8(v8bf lo, v8bf hi) {
    return __builtin_shufflevector(lo, hi, 0,1,2,3,4,5,6,7,8,9,10,11,12,13,14,15);
}
static __device__ __forceinline__ v8f zero8() {
    v8f z = {0.f,0.f,0.f,0.f,0.f,0.f,0.f,0.f};
    return z;
}
static __device__ __forceinline__ v8f wmma_bf16(v16bf a, v16bf b, v8f c) {
    return __builtin_amdgcn_wmma_f32_16x16x32_bf16(
        /*neg_a=*/false, a, /*neg_b=*/false, b,
        /*c_mod=*/(short)0, c, /*reuse_a=*/false, /*reuse_b=*/false);
}

// ---------------------------------------------------------------------------
// fp32 -> bf16 conversion (scale folded in: 1/sqrt(d) goes into Wq)
// ---------------------------------------------------------------------------
__global__ __launch_bounds__(256) void cvt_bf16(const float* __restrict__ src,
                                                bf16* __restrict__ dst,
                                                int n, float scale) {
    int i = (blockIdx.x * 256 + threadIdx.x) * 4;
    if (i + 3 < n) {
        v4f v = *(const v4f*)(src + i);
        v4bf o;
        o[0] = (bf16)(v[0] * scale);
        o[1] = (bf16)(v[1] * scale);
        o[2] = (bf16)(v[2] * scale);
        o[3] = (bf16)(v[3] * scale);
        *(v4bf*)(dst + i) = o;
    }
}

// ---------------------------------------------------------------------------
// C[4096,1024](bf16) = A[4096,1024](bf16) @ B[1024,1024](bf16)
// wave tile: 32M x 64N (2 A-frags shared across 4 B-frags -> 8 WMMA / 4 A + 64 B loads)
// block: 8 waves (2x4) = 64M x 256N.
// ---------------------------------------------------------------------------
__global__ __launch_bounds__(256) void qkv_gemm(const bf16* __restrict__ A,
                                                const bf16* __restrict__ B,
                                                bf16* __restrict__ C) {
    const int tid  = threadIdx.x;
    const int lane = tid & 31;
    const int wave = tid >> 5;
    const int lm = lane & 15;     // M (A) or N (B/C) sub-index
    const int lh = lane >> 4;     // half-lane select
    const int m0 = blockIdx.x * 64 + (wave >> 2) * 32;
    const int n0 = blockIdx.y * 256 + (wave & 3) * 64;

    v8f acc[2][4];
    #pragma unroll
    for (int mi = 0; mi < 2; ++mi)
        #pragma unroll
        for (int nt = 0; nt < 4; ++nt) acc[mi][nt] = zero8();

    const bf16* arow0 = A + (size_t)(m0 + lm) * DMODEL;
    const bf16* arow1 = A + (size_t)(m0 + 16 + lm) * DMODEL;
    for (int kb = 0; kb < DMODEL; kb += 32) {
        v16bf a0 = cat8(*(const v8bf*)(arow0 + kb + lh * 8),
                        *(const v8bf*)(arow0 + kb + 16 + lh * 8));
        v16bf a1 = cat8(*(const v8bf*)(arow1 + kb + lh * 8),
                        *(const v8bf*)(arow1 + kb + 16 + lh * 8));
        const bf16* brow = B + (size_t)kb * DMODEL + n0 + lm;
        #pragma unroll
        for (int nt = 0; nt < 4; ++nt) {
            v16bf bfrag;
            #pragma unroll
            for (int s = 0; s < 16; ++s)
                bfrag[s] = brow[(size_t)((s & 7) + 8 * lh + ((s >> 3) << 4)) * DMODEL + nt * 16];
            acc[0][nt] = wmma_bf16(a0, bfrag, acc[0][nt]);
            acc[1][nt] = wmma_bf16(a1, bfrag, acc[1][nt]);
        }
    }
    #pragma unroll
    for (int mi = 0; mi < 2; ++mi) {
        #pragma unroll
        for (int nt = 0; nt < 4; ++nt) {
            const int col = n0 + nt * 16 + lm;
            #pragma unroll
            for (int r = 0; r < 8; ++r) {
                const int row = m0 + mi * 16 + r + 8 * lh;
                C[(size_t)row * DMODEL + col] = (bf16)acc[mi][nt][r];
            }
        }
    }
}

// ---------------------------------------------------------------------------
// Flash attention: 16 queries per block, 8 waves x 128 output columns.
// S^T = K @ Q^T via WMMA (Q transposed in LDS), cross-wave ds_add_f32
// reduction, online softmax on wave 0 (32 lanes, shfl_xor pair-combine),
// O += P @ V via WMMA.
// ---------------------------------------------------------------------------
__global__ __launch_bounds__(256) void attn(const bf16* __restrict__ Qb,
                                            const bf16* __restrict__ Kb,
                                            const bf16* __restrict__ Vb,
                                            float* __restrict__ Out) {
    __shared__ bf16  Qt[DMODEL * 16];     // Q^T [feat][query], 32 KB
    __shared__ float Ssc[16 * 33];        // scores [query][key], pad stride 33
    __shared__ bf16  Pbuf[16 * 40];       // P bf16 [query][key], pad stride 40
    __shared__ float rbuf[16];            // per-row alpha / 1/l

    const int tid  = threadIdx.x;
    const int lane = tid & 31;
    const int wave = tid >> 5;
    const int lm = lane & 15;
    const int lh = lane >> 4;
    const int q0   = blockIdx.x * 16;
    const int col0 = wave * 128;          // this wave's feature slice

    // Stage Q transposed into LDS (bf16).
    for (int c = tid; c < 16 * 128; c += 256) {
        const int q  = c & 15;
        const int f0 = (c >> 4) << 3;
        const bf16* src = Qb + (size_t)(q0 + q) * DMODEL + f0;
        #pragma unroll
        for (int i = 0; i < 8; ++i) Qt[(f0 + i) * 16 + q] = src[i];
    }
    for (int c = tid; c < 16 * 33; c += 256) Ssc[c] = 0.f;
    __syncthreads();

    v8f o[8];
    #pragma unroll
    for (int t = 0; t < 8; ++t) o[t] = zero8();
    float mrun = -3.0e38f, lrun = 0.f;

    for (int kb2 = 0; kb2 < SEQ; kb2 += 32) {
        // ---- Phase A: partial S^T = K_tile @ Q^T over this wave's 128 feats
        v8f s0 = zero8(), s1 = zero8();
        #pragma unroll
        for (int ks = 0; ks < 4; ++ks) {
            const int kf = col0 + ks * 32;
            const bf16* qrow = &Qt[kf * 16 + lm];
            v16bf bq;
            #pragma unroll
            for (int s = 0; s < 16; ++s)
                bq[s] = qrow[((s & 7) + 8 * lh + ((s >> 3) << 4)) * 16];
            const bf16* k0r = Kb + (size_t)(kb2 + lm) * DMODEL + kf;
            const bf16* k1r = Kb + (size_t)(kb2 + 16 + lm) * DMODEL + kf;
            v16bf a0 = cat8(*(const v8bf*)(k0r + lh * 8),
                            *(const v8bf*)(k0r + 16 + lh * 8));
            v16bf a1 = cat8(*(const v8bf*)(k1r + lh * 8),
                            *(const v8bf*)(k1r + 16 + lh * 8));
            s0 = wmma_bf16(a0, bq, s0);
            s1 = wmma_bf16(a1, bq, s1);
        }
        // reduce partials across waves (C layout: lane=query, vgpr r = key)
        #pragma unroll
        for (int r = 0; r < 8; ++r) {
            atomicAdd(&Ssc[lm * 33 + (r + 8 * lh)], s0[r]);
            atomicAdd(&Ssc[lm * 33 + 16 + (r + 8 * lh)], s1[r]);
        }
        __syncthreads();

        // ---- Phase B: online softmax, wave 0, 2 lanes per query row
        //      (lane q handles keys 0..15, lane q+16 handles keys 16..31)
        if (wave == 0) {
            const int q  = lm;
            const int kbase = lh * 16;
            float sv[16];
            float tmax = -3.0e38f;
            #pragma unroll
            for (int k = 0; k < 16; ++k) {
                sv[k] = Ssc[q * 33 + kbase + k];
                Ssc[q * 33 + kbase + k] = 0.f;       // ready for next tile
                tmax = fmaxf(tmax, sv[k]);
            }
            tmax = fmaxf(tmax, __shfl_xor(tmax, 16, 32));   // pair-combine max
            const float mnew  = fmaxf(mrun, tmax);
            const float alpha = __expf(mrun - mnew);
            float psum = 0.f;
            #pragma unroll
            for (int k = 0; k < 16; ++k) {
                const float p = __expf(sv[k] - mnew);
                psum += p;
                Pbuf[q * 40 + kbase + k] = (bf16)p;
            }
            psum += __shfl_xor(psum, 16, 32);               // pair-combine sum
            lrun = lrun * alpha + psum;   // replicated identically in both lanes
            mrun = mnew;
            if (lane < 16) rbuf[q] = alpha;
        }
        __syncthreads();

        // ---- Phase C: rescale O by alpha, O += P @ V_tile
        v4f a0 = *(const v4f*)&rbuf[lh * 8];
        v4f a1 = *(const v4f*)&rbuf[lh * 8 + 4];
        float alr[8] = {a0[0], a0[1], a0[2], a0[3], a1[0], a1[1], a1[2], a1[3]};
        #pragma unroll
        for (int t = 0; t < 8; ++t)
            #pragma unroll
            for (int r = 0; r < 8; ++r) o[t][r] *= alr[r];

        const bf16* prow = &Pbuf[lm * 40];
        v16bf ap = cat8(*(const v8bf*)(prow + lh * 8),
                        *(const v8bf*)(prow + 16 + lh * 8));
        const bf16* vrow = Vb + (size_t)kb2 * DMODEL + col0 + lm;
        #pragma unroll
        for (int nt = 0; nt < 8; ++nt) {
            v16bf bv;
            #pragma unroll
            for (int s = 0; s < 16; ++s)
                bv[s] = vrow[(size_t)((s & 7) + 8 * lh + ((s >> 3) << 4)) * DMODEL + nt * 16];
            o[nt] = wmma_bf16(ap, bv, o[nt]);
        }
        if (kb2 + 32 < SEQ) {   // prefetch next K/V tile rows (L2-resident)
            __builtin_prefetch(Kb + (size_t)(kb2 + 32 + lm) * DMODEL + col0, 0, 1);
            __builtin_prefetch(Vb + (size_t)(kb2 + 32 + lm + 16 * lh) * DMODEL + col0, 0, 1);
        }
    }

    // ---- Finalize: divide by softmax denominator, store fp32
    if (wave == 0 && lane < 16) rbuf[lane] = 1.0f / lrun;
    __syncthreads();
    v4f l0 = *(const v4f*)&rbuf[lh * 8];
    v4f l1 = *(const v4f*)&rbuf[lh * 8 + 4];
    float lir[8] = {l0[0], l0[1], l0[2], l0[3], l1[0], l1[1], l1[2], l1[3]};
    #pragma unroll
    for (int nt = 0; nt < 8; ++nt) {
        const int col = col0 + nt * 16 + lm;
        #pragma unroll
        for (int r = 0; r < 8; ++r) {
            const int row = q0 + r + 8 * lh;
            Out[(size_t)row * DMODEL + col] = o[nt][r] * lir[r];
        }
    }
}

// ---------------------------------------------------------------------------
extern "C" void kernel_launch(void* const* d_in, const int* in_sizes, int n_in,
                              void* d_out, int out_size, void* d_ws, size_t ws_size,
                              hipStream_t stream) {
    const float* x  = (const float*)d_in[0];
    const float* wq = (const float*)d_in[1];
    const float* wk = (const float*)d_in[2];
    const float* wv = (const float*)d_in[3];
    float* out = (float*)d_out;

    // Workspace layout (bf16), 38 MB total.
    char* ws = (char*)d_ws;
    bf16* Xb  = (bf16*)(ws);                          // 8 MB
    bf16* Wqb = (bf16*)(ws + (size_t)8  * (1 << 20)); // 2 MB
    bf16* Wkb = (bf16*)(ws + (size_t)10 * (1 << 20)); // 2 MB
    bf16* Wvb = (bf16*)(ws + (size_t)12 * (1 << 20)); // 2 MB
    bf16* Qb  = (bf16*)(ws + (size_t)14 * (1 << 20)); // 8 MB
    bf16* Kb  = (bf16*)(ws + (size_t)22 * (1 << 20)); // 8 MB
    bf16* Vb  = (bf16*)(ws + (size_t)30 * (1 << 20)); // 8 MB

    const int nx = SEQ * DMODEL;
    const int nw = DMODEL * DMODEL;
    const float qscale = 0.03125f;  // 1/sqrt(1024), folded into Wq

    cvt_bf16<<<nx / 1024, 256, 0, stream>>>(x,  Xb,  nx, 1.0f);
    cvt_bf16<<<nw / 1024, 256, 0, stream>>>(wq, Wqb, nw, qscale);
    cvt_bf16<<<nw / 1024, 256, 0, stream>>>(wk, Wkb, nw, 1.0f);
    cvt_bf16<<<nw / 1024, 256, 0, stream>>>(wv, Wvb, nw, 1.0f);

    dim3 ggemm(SEQ / 64, DMODEL / 256);
    qkv_gemm<<<ggemm, 256, 0, stream>>>(Xb, Wqb, Qb);
    qkv_gemm<<<ggemm, 256, 0, stream>>>(Xb, Wkb, Kb);
    qkv_gemm<<<ggemm, 256, 0, stream>>>(Xb, Wvb, Vb);

    attn<<<SEQ / 16, 256, 0, stream>>>(Qb, Kb, Vb, out);
}
